// AudioTokenizer_73933567033599
// MI455X (gfx1250) — compile-verified, matrix operand order checked by reading, not run on hardware
//
#include <hip/hip_runtime.h>
#include <hip/hip_bf16.h>
#include <cstdint>

// ---------------------------------------------------------------------------
// CDNA5 / gfx1250 implementation of the audio tokenizer.
//  - fp32 VALU direct convs for the encoder front-end
//  - v_wmma_f32_16x16x32_f16 tiled GEMMs for LSTM input projections and RVQ
//  - persistent wave32 LSTM recurrence kernel with software grid barrier
// ---------------------------------------------------------------------------

typedef __attribute__((ext_vector_type(16))) _Float16 v16h;
typedef __attribute__((ext_vector_type(8)))  float    v8f;

#define TPB 256
#define LSTM_GRID 8

__device__ __forceinline__ float sigmoidf_(float x) { return 1.f / (1.f + __expf(-x)); }
__device__ __forceinline__ float eluf_(float x)     { return x > 0.f ? x : (__expf(x) - 1.f); }

// ---------------------------------------------------------------------------
// Generic direct Conv1d: out[b,co,t] = bias[co] + sum_ci,k w[co,ci,k]*in[b,ci,t*s+k-pad]
// One thread per output element; consecutive threads share co -> coalesced on T.
// ---------------------------------------------------------------------------
__global__ __launch_bounds__(TPB) void conv1d_k(
    const float* __restrict__ in, const float* __restrict__ w,
    const float* __restrict__ bias, float* __restrict__ out,
    int Cin, int Cout, int Tin, int Tout,
    int K, int stride, int pad, int do_elu, long long total)
{
  long long idx = (long long)blockIdx.x * TPB + threadIdx.x;
  if (idx >= total) return;
  int t  = (int)(idx % Tout);
  int co = (int)((idx / Tout) % Cout);
  int b  = (int)(idx / ((long long)Tout * Cout));
  float acc = bias[co];
  const float* wb = w  + (long long)co * Cin * K;
  const float* ib = in + (long long)b  * Cin * Tin;
  int tb = t * stride - pad;
  for (int ci = 0; ci < Cin; ++ci) {
    const float* ip = ib + (long long)ci * Tin;
    const float* wp = wb + ci * K;
    for (int k = 0; k < K; ++k) {
      int tt = tb + k;
      if (tt >= 0 && tt < Tin) acc = fmaf(wp[k], ip[tt], acc);
    }
  }
  if (do_elu) acc = eluf_(acc);
  out[idx] = acc;
}

// ---------------------------------------------------------------------------
// Fused: y[b,co,t] = conv3(r1)@(t*s) + down_conv(h)@t + biases
// (residual conv2 evaluated only at strided positions; skips storing r2)
// ---------------------------------------------------------------------------
__global__ __launch_bounds__(TPB) void resdown_k(
    const float* __restrict__ r1, const float* __restrict__ h,
    const float* __restrict__ w2, const float* __restrict__ b2,
    const float* __restrict__ wd, const float* __restrict__ bd,
    float* __restrict__ y,
    int Cin, int Cout, int Tin, int Tsc, int s, long long total)
{
  long long idx = (long long)blockIdx.x * TPB + threadIdx.x;
  if (idx >= total) return;
  int t  = (int)(idx % Tsc);
  int co = (int)((idx / Tsc) % Cout);
  int b  = (int)(idx / ((long long)Tsc * Cout));
  float acc = b2[co] + bd[co];
  int p = t * s;
  {   // residual conv2: K=3, pad=1, evaluated at position p
    const float* wb = w2 + (long long)co * Cout * 3;
    const float* rb = r1 + (long long)b  * Cout * Tin;
    for (int ci = 0; ci < Cout; ++ci) {
      const float* ip = rb + (long long)ci * Tin;
      const float* wp = wb + ci * 3;
      for (int k = 0; k < 3; ++k) {
        int tt = p + k - 1;
        if (tt >= 0 && tt < Tin) acc = fmaf(wp[k], ip[tt], acc);
      }
    }
  }
  {   // downsample conv: K=2s, stride=s, pad=s/2
    int Kd = 2 * s, pd = s >> 1;
    const float* wb = wd + (long long)co * Cin * Kd;
    const float* hb = h  + (long long)b  * Cin * Tin;
    for (int ci = 0; ci < Cin; ++ci) {
      const float* ip = hb + (long long)ci * Tin;
      const float* wp = wb + ci * Kd;
      for (int k = 0; k < Kd; ++k) {
        int tt = p + k - pd;
        if (tt >= 0 && tt < Tin) acc = fmaf(wp[k], ip[tt], acc);
      }
    }
  }
  y[idx] = acc;
}

// weight_norm (dim=0): wn[co][ci] = g[co] * v[co][ci] / ||v[co,:]||
__global__ __launch_bounds__(TPB) void wnorm_k(
    const float* __restrict__ v, const float* __restrict__ g,
    float* __restrict__ wn, int cout)
{
  int co = blockIdx.x * TPB + threadIdx.x;
  if (co >= cout) return;
  const float* vr = v + (long long)co * cout;
  float s = 0.f;
  for (int ci = 0; ci < cout; ++ci) s = fmaf(vr[ci], vr[ci], s);
  float sc = g[co] * rsqrtf(s);
  float* wr = wn + (long long)co * cout;
  for (int ci = 0; ci < cout; ++ci) wr[ci] = vr[ci] * sc;
}

// (B,C,T) -> (B*T, C)
__global__ __launch_bounds__(TPB) void nct_to_ntc_k(
    const float* __restrict__ in, float* __restrict__ out,
    int C, int T, long long total)
{
  long long idx = (long long)blockIdx.x * TPB + threadIdx.x;
  if (idx >= total) return;
  int c = (int)(idx % C);
  int t = (int)((idx / C) % T);
  int b = (int)(idx / ((long long)C * T));
  out[idx] = in[((long long)b * C + c) * T + t];
}

// (B*T, C) -> (B,C,T)
__global__ __launch_bounds__(TPB) void ntc_to_nct_k(
    const float* __restrict__ in, float* __restrict__ out,
    int C, int T, long long total)
{
  long long idx = (long long)blockIdx.x * TPB + threadIdx.x;
  if (idx >= total) return;
  int t = (int)(idx % T);
  int c = (int)((idx / T) % C);
  int b = (int)(idx / ((long long)C * T));
  out[idx] = in[((long long)b * T + t) * C + c];
}

__global__ __launch_bounds__(TPB) void cvt_f16_k(
    const float* __restrict__ in, _Float16* __restrict__ out, long long n)
{
  long long i = (long long)blockIdx.x * TPB + threadIdx.x;
  if (i < n) out[i] = (_Float16)in[i];
}

__global__ __launch_bounds__(TPB) void addv_k(
    const float* __restrict__ a, const float* __restrict__ b,
    float* __restrict__ o, int n)
{
  int i = blockIdx.x * TPB + threadIdx.x;
  if (i < n) o[i] = a[i] + b[i];
}

// ---------------------------------------------------------------------------
// WMMA GEMM: C[M16 x N] = A[M16 x K](f16, row-major) * B[N x K](f16, row-major)^T + bias
// One 16x16 C-tile per wave32; K in chunks of 32 via v_wmma_f32_16x16x32_f16.
// Operand packing follows ISA 7.12.2: lane half kh = lane>>4, VGPR v holds the
// K pair ((v&3)*2 + (v>>2)*16 + kh*8, +1)  -> one b32 load per VGPR.
// M is padded to a multiple of 16 in workspace, so no edge guards are needed.
// ---------------------------------------------------------------------------
__global__ __launch_bounds__(TPB) void wmma_gemm_k(
    const _Float16* __restrict__ A, const _Float16* __restrict__ Bw,
    const float* __restrict__ bias, float* __restrict__ C,
    int Mt, int Nt, int K, int N)
{
  int wave = threadIdx.x >> 5;
  int lane = threadIdx.x & 31;
  int tile = blockIdx.x * (TPB / 32) + wave;
  if (tile >= Mt * Nt) return;               // uniform per wave: EXEC stays all-1s
  int tm = tile / Nt, tn = tile % Nt;
  int mr = lane & 15, kh = lane >> 4;
  const _Float16* Ap = A  + (long long)(tm * 16 + mr) * K;
  const _Float16* Bp = Bw + (long long)(tn * 16 + mr) * K;
  v8f acc = {};
  for (int kk = 0; kk < K; kk += 32) {
    union { v16h h; uint32_t u[8]; } a, bm;
#pragma unroll
    for (int vv = 0; vv < 8; ++vv) {
      int koff = kk + ((vv & 3) << 1) + ((vv >> 2) << 4) + (kh << 3);
      a.u[vv]  = *(const uint32_t*)(Ap + koff);
      bm.u[vv] = *(const uint32_t*)(Bp + koff);
    }
    acc = __builtin_amdgcn_wmma_f32_16x16x32_f16(
        false, a.h, false, bm.h, (short)0, acc, false, false);
  }
  int n = lane & 15, half = lane >> 4;
  float bb = bias ? bias[tn * 16 + n] : 0.f;
#pragma unroll
  for (int vv = 0; vv < 8; ++vv) {            // C/D layout: VGPR vv -> M = vv + 8*half
    int m = tm * 16 + vv + (half << 3);
    C[(long long)m * N + tn * 16 + n] = acc[vv] + bb;
  }
}

// ---------------------------------------------------------------------------
// Persistent LSTM recurrence. 8 blocks x 256 threads = 2048 threads = 4*512 gate
// rows. Block 'blk' owns hidden units j in [blk*64, blk*64+64) for all 4 gates
// and both batch rows, so the c/h update is block-local (c kept in registers).
// One software grid barrier per timestep; h state lives in global memory.
// Gin = x_t @ W_ih^T + (b_ih+b_hh), precomputed by the WMMA GEMM (rows b*T+t).
// ---------------------------------------------------------------------------
__global__ __launch_bounds__(TPB) void lstm_recur_k(
    const float* __restrict__ Gin, const float* __restrict__ whh,
    float* __restrict__ hstate, float* __restrict__ Xout,
    int T, int* __restrict__ bar_cnt, int* __restrict__ bar_gen)
{
  __shared__ float h_s[1024];   // h for b=0 (0..511) and b=1 (512..1023)
  __shared__ float g_s[512];    // gates[4][2][64]
  const int tid  = threadIdx.x;
  const int blk  = blockIdx.x;
  const int gate = tid >> 6;
  const int jl   = tid & 63;
  const int row  = gate * 512 + blk * 64 + jl;
  const float* wrow = whh + (long long)row * 512;
  float c_reg = 0.f;            // used by updater threads (tid < 128)
  int lgen = 0;
  for (int t = 0; t < T; ++t) {
    for (int i = tid; i < 1024; i += TPB) h_s[i] = hstate[i];
    __syncthreads();
    float a0 = 0.f, a1 = 0.f;
#pragma unroll 4
    for (int d = 0; d < 512; ++d) {
      float w = wrow[d];
      a0 = fmaf(w, h_s[d], a0);
      a1 = fmaf(w, h_s[512 + d], a1);
    }
    g_s[gate * 128 + jl]      = Gin[(long long)t * 2048 + row]       + a0;  // b=0
    g_s[gate * 128 + 64 + jl] = Gin[(long long)(T + t) * 2048 + row] + a1;  // b=1
    __syncthreads();
    if (tid < 128) {
      int b = tid >> 6, j = tid & 63;
      float gi = g_s[      b * 64 + j];
      float gf = g_s[128 + b * 64 + j];
      float gg = g_s[256 + b * 64 + j];
      float go = g_s[384 + b * 64 + j];
      c_reg = sigmoidf_(gf) * c_reg + sigmoidf_(gi) * tanhf(gg);
      float hn = sigmoidf_(go) * tanhf(c_reg);
      hstate[b * 512 + blk * 64 + j] = hn;
      Xout[((long long)b * T + t) * 512 + blk * 64 + j] = hn;
    }
    __threadfence();
    // --- software grid barrier (sense via generation counter) ---
    __syncthreads();
    if (tid == 0) {
      if (atomicAdd(bar_cnt, 1) == LSTM_GRID - 1) {
        *bar_cnt = 0;
        __threadfence();
        atomicAdd(bar_gen, 1);
      } else {
        volatile int* vg = bar_gen;
        while (*vg <= lgen) { __builtin_amdgcn_s_sleep(1); }
      }
      ++lgen;
      __threadfence();
    }
    __syncthreads();
  }
}

// Row-wise argmax over codebook logits; first-index tie-break (jnp.argmax).
__global__ __launch_bounds__(TPB) void argmax_k(
    const float* __restrict__ logits, int* __restrict__ idx_out,
    float* __restrict__ fidx_out, int CBSz, int cb, int ncb)
{
  int m = blockIdx.x;
  const float* rowp = logits + (long long)m * CBSz;
  __shared__ float sv[TPB];
  __shared__ int   si[TPB];
  float best = -3.4e38f; int bi = 0;
  for (int j = threadIdx.x; j < CBSz; j += TPB) {
    float v = rowp[j];
    if (v > best) { best = v; bi = j; }
  }
  sv[threadIdx.x] = best; si[threadIdx.x] = bi;
  __syncthreads();
  for (int off = TPB / 2; off > 0; off >>= 1) {
    if (threadIdx.x < off) {
      float ov = sv[threadIdx.x + off]; int oi = si[threadIdx.x + off];
      if (ov > sv[threadIdx.x] || (ov == sv[threadIdx.x] && oi < si[threadIdx.x])) {
        sv[threadIdx.x] = ov; si[threadIdx.x] = oi;
      }
    }
    __syncthreads();
  }
  if (threadIdx.x == 0) {
    idx_out[m] = si[0];
    fidx_out[(long long)m * ncb + cb] = (float)si[0];
  }
}

// res -= codebook[idx];  quant (in d_out) = first ? q : quant + q
__global__ __launch_bounds__(TPB) void rvq_update_k(
    const int* __restrict__ idx, const float* __restrict__ cbk,
    float* __restrict__ res, float* __restrict__ quant,
    int D, int first, long long total)
{
  long long i = (long long)blockIdx.x * TPB + threadIdx.x;
  if (i >= total) return;
  int d = (int)(i % D);
  long long m = i / D;
  float q = cbk[(long long)idx[m] * D + d];
  res[i] -= q;
  quant[i] = first ? q : (quant[i] + q);
}

// ---------------------------------------------------------------------------
extern "C" void kernel_launch(void* const* d_in, const int* in_sizes, int n_in,
                              void* d_out, int out_size, void* d_ws, size_t ws_size,
                              hipStream_t stream) {
  (void)in_sizes; (void)n_in; (void)out_size; (void)ws_size;

  // ---- input pointer map (JAX sorted-key flatten of setup_inputs dict) ----
  // params/blocks[i]: 0 down_b, 1 down_w, 2 norm_b, 3 norm_g, 4 norm_v,
  //                   5 res_b1, 6 res_b2, 7 res_w1, 8 res_w2   (i*9 + j)
  // 36..43 codebooks, 44 final_b, 45 final_w, 46 init_b, 47 init_w,
  // 48..55 lstm[l]{b_hh,b_ih,w_hh,w_ih}, 56..63 proj_b, 64..71 proj_w, 72 x
  auto BP = [&](int blk, int j) { return (const float*)d_in[blk * 9 + j]; };
  const float* x       = (const float*)d_in[72];
  const float* final_b = (const float*)d_in[44];
  const float* final_w = (const float*)d_in[45];
  const float* init_b  = (const float*)d_in[46];
  const float* init_w  = (const float*)d_in[47];

  // ---- temporal sizes through the encoder ----
  int Ts[5]; Ts[0] = 64000;
  const int cins[4]  = {64, 128, 256, 512};
  const int couts[4] = {128, 256, 512, 512};
  const int strd[4]  = {2, 4, 5, 8};
  for (int i = 0; i < 4; ++i) {
    int s = strd[i];
    Ts[i + 1] = (Ts[i] + 2 * (s / 2) - 2 * s) / s + 1;   // 32000, 8000, 1599, 199
  }
  const int Tf  = Ts[4];
  const int M   = 2 * Tf;                 // B*T rows
  const int M16 = (M + 15) & ~15;         // padded for guard-free WMMA tiles
  const int Mt  = M16 / 16;

  // ---- workspace carving ----
  char* wsb = (char*)d_ws;
  size_t off = 0;
  auto carve = [&](size_t bytes) -> void* {
    void* p = wsb + off;
    off += (bytes + 255) & ~(size_t)255;
    return p;
  };
  float*    bufH   = (float*)carve((size_t)2 * 64  * 64000 * 4);  // h (max 8.19M fl)
  float*    bufR   = (float*)carve((size_t)2 * 256 * 32000 * 4);  // r1 (max 16.38M fl)
  float*    bufY   = (float*)carve((size_t)2 * 128 * 32000 * 4);  // y (max 8.19M fl)
  float*    wnW    = (float*)carve((size_t)512 * 512 * 4);
  float*    Xa     = (float*)carve((size_t)M16 * 512 * 4);
  float*    Xb     = (float*)carve((size_t)M16 * 512 * 4);
  float*    resF   = (float*)carve((size_t)M16 * 512 * 4);
  _Float16* Xh     = (_Float16*)carve((size_t)M16 * 512 * 2);
  float*    Gin    = (float*)carve((size_t)M16 * 2048 * 4);
  _Float16* Wh     = (_Float16*)carve((size_t)2048 * 512 * 2);
  float*    biasC  = (float*)carve((size_t)2048 * 4);
  float*    logits = (float*)carve((size_t)M16 * 1024 * 4);
  int*      idxBuf = (int*)carve((size_t)M * 4);
  float*    hstate = (float*)carve((size_t)1024 * 4);
  int*      bar    = (int*)carve(256);

  auto nblk = [](long long total) { return (unsigned)((total + TPB - 1) / TPB); };

  // ---- stem: h = elu(conv1d(x, k7, p3)) ----
  {
    long long tot = (long long)2 * 64 * Ts[0];
    conv1d_k<<<nblk(tot), TPB, 0, stream>>>(x, init_w, init_b, bufH,
                                            1, 64, Ts[0], Ts[0], 7, 1, 3, 1, tot);
  }

  // ---- 4 residual-downsample blocks ----
  for (int i = 0; i < 4; ++i) {
    int cin = cins[i], cout = couts[i], s = strd[i], Tin = Ts[i], Tsc = Ts[i + 1];
    long long t1 = (long long)2 * cout * Tin;
    conv1d_k<<<nblk(t1), TPB, 0, stream>>>(bufH, BP(i, 7), BP(i, 5), bufR,
                                           cin, cout, Tin, Tin, 3, 1, 1, 1, t1);
    long long t2 = (long long)2 * cout * Tsc;
    resdown_k<<<nblk(t2), TPB, 0, stream>>>(bufR, bufH, BP(i, 8), BP(i, 6),
                                            BP(i, 1), BP(i, 0), bufY,
                                            cin, cout, Tin, Tsc, s, t2);
    wnorm_k<<<(cout + TPB - 1) / TPB, TPB, 0, stream>>>(BP(i, 4), BP(i, 3), wnW, cout);
    conv1d_k<<<nblk(t2), TPB, 0, stream>>>(bufY, wnW, BP(i, 2), bufH,
                                           cout, cout, Tsc, Tsc, 1, 1, 0, 1, t2);
  }

  // ---- LSTM (2 layers): WMMA input projection + persistent recurrence ----
  long long tM = (long long)M * 512;
  nct_to_ntc_k<<<nblk(tM), TPB, 0, stream>>>(bufH, Xa, 512, Tf, tM);
  float* Xin = Xa; float* Xout = Xb;
  for (int l = 0; l < 2; ++l) {
    const float* b_hh = (const float*)d_in[48 + 4 * l + 0];
    const float* b_ih = (const float*)d_in[48 + 4 * l + 1];
    const float* w_hh = (const float*)d_in[48 + 4 * l + 2];
    const float* w_ih = (const float*)d_in[48 + 4 * l + 3];
    cvt_f16_k<<<nblk((long long)2048 * 512), TPB, 0, stream>>>(w_ih, Wh, (long long)2048 * 512);
    addv_k<<<nblk(2048), TPB, 0, stream>>>(b_ih, b_hh, biasC, 2048);
    cvt_f16_k<<<nblk((long long)M16 * 512), TPB, 0, stream>>>(Xin, Xh, (long long)M16 * 512);
    int tiles = Mt * (2048 / 16);
    wmma_gemm_k<<<(tiles + 7) / 8, TPB, 0, stream>>>(Xh, Wh, biasC, Gin, Mt, 2048 / 16, 512, 2048);
    hipMemsetAsync(hstate, 0, 1024 * 4, stream);
    hipMemsetAsync(bar, 0, 8, stream);
    lstm_recur_k<<<LSTM_GRID, TPB, 0, stream>>>(Gin, w_hh, hstate, Xout, Tf, bar, bar + 1);
    float* tmp = Xin; Xin = Xout; Xout = tmp;
  }

  // ---- final conv (k7, p3) ----
  ntc_to_nct_k<<<nblk(tM), TPB, 0, stream>>>(Xin, bufY, 512, Tf, tM);
  conv1d_k<<<nblk(tM), TPB, 0, stream>>>(bufY, final_w, final_b, bufH,
                                         512, 512, Tf, Tf, 7, 1, 3, 0, tM);
  nct_to_ntc_k<<<nblk(tM), TPB, 0, stream>>>(bufH, resF, 512, Tf, tM);

  // ---- residual VQ: 8 sequential codebooks, WMMA logits each round ----
  float* out_idx = (float*)d_out;                 // (B,T,8) indices as float
  float* out_q   = out_idx + (size_t)M * 8;       // (B,T,512) quantized
  for (int cb = 0; cb < 8; ++cb) {
    cvt_f16_k<<<nblk((long long)M16 * 512), TPB, 0, stream>>>(resF, Xh, (long long)M16 * 512);
    cvt_f16_k<<<nblk((long long)1024 * 512), TPB, 0, stream>>>(
        (const float*)d_in[64 + cb], Wh, (long long)1024 * 512);
    int tiles = Mt * (1024 / 16);
    wmma_gemm_k<<<(tiles + 7) / 8, TPB, 0, stream>>>(
        Xh, Wh, (const float*)d_in[56 + cb], logits, Mt, 1024 / 16, 512, 1024);
    argmax_k<<<M, TPB, 0, stream>>>(logits, idxBuf, out_idx, 1024, cb, 8);
    long long tu = (long long)M * 512;
    rvq_update_k<<<nblk(tu), TPB, 0, stream>>>(idxBuf, (const float*)d_in[36 + cb],
                                               resF, out_q, 512, cb == 0 ? 1 : 0, tu);
  }
}
